// LatticeNNConv_39926015984149
// MI455X (gfx1250) — matrix operand order repaired and checked by reading, not compiled, not run on hardware
//
#include <hip/hip_runtime.h>
#include <math.h>

// ---------------------------------------------------------------------------
// CDNA5 (gfx1250) fused NNConv GNN.
// Heavy math on v_wmma_f32_16x16x32_bf16 (K=32 matches the hidden size).
// ---------------------------------------------------------------------------

typedef __bf16 v16bf __attribute__((ext_vector_type(16)));
typedef float  v8f   __attribute__((ext_vector_type(8)));

union Frag16 {            // 16x32 (A) or 32x16 (B) bf16 fragment, 8 VGPRs
    v16bf v;
    unsigned int u[8];
};

static __device__ __forceinline__ v8f wmma_bf16(v16bf a, v16bf b, v8f c) {
    // 8 args: (neg_a, A, neg_b, B, c_mod, C, reuse_a, reuse_b)
    return __builtin_amdgcn_wmma_f32_16x16x32_bf16(false, a, false, b,
                                                   (short)0, c, false, false);
}

static __device__ __forceinline__ unsigned short f2bf(float f) {
    union { float f; unsigned int u; } x; x.f = f;
    unsigned int r = x.u + 0x7FFFu + ((x.u >> 16) & 1u);   // RNE
    return (unsigned short)(r >> 16);
}
static __device__ __forceinline__ unsigned int pack_bf(float lo, float hi) {
    return ((unsigned int)f2bf(hi) << 16) | (unsigned int)f2bf(lo);
}
static __device__ __forceinline__ float sigmoidf(float x) {
    return 1.0f / (1.0f + __expf(-x));
}

#define H 32

// ---------------------------------------------------------------------------
// Edge features + 3-layer edge MLP -> e3 (bf16, [E,32]) ; degree count.
// ---------------------------------------------------------------------------
__global__ void edgenet_kernel(const int* __restrict__ eidx,
                               const float* __restrict__ coords,
                               const float* __restrict__ w1, const float* __restrict__ b1,
                               const float* __restrict__ w2, const float* __restrict__ b2,
                               const float* __restrict__ w3, const float* __restrict__ b3,
                               unsigned short* __restrict__ e3,
                               float* __restrict__ deg, int E) {
    int e = blockIdx.x * blockDim.x + threadIdx.x;
    if (e >= E) return;
    int s = eidx[e], d = eidx[E + e];
    float t0 = coords[d * 2 + 0] - coords[s * 2 + 0];
    float t1 = coords[d * 2 + 1] - coords[s * 2 + 1];
    float L  = sqrtf(t0 * t0 + t1 * t1);
    float ef[4] = { t0, t1, t0 / L, L };

    float a1[8];
#pragma unroll
    for (int j = 0; j < 8; ++j) {
        float acc = b1[j];
#pragma unroll
        for (int k = 0; k < 4; ++k) acc += ef[k] * w1[k * 8 + j];
        a1[j] = fmaxf(acc, 0.f);
    }
    float a2[16];
#pragma unroll
    for (int j = 0; j < 16; ++j) {
        float acc = b2[j];
#pragma unroll
        for (int k = 0; k < 8; ++k) acc += a1[k] * w2[k * 16 + j];
        a2[j] = fmaxf(acc, 0.f);
    }
#pragma unroll
    for (int o = 0; o < 32; ++o) {
        float acc = b3[o];
#pragma unroll
        for (int k = 0; k < 16; ++k) acc += a2[k] * w3[k * 32 + o];
        e3[(size_t)e * 32 + o] = f2bf(fmaxf(acc, 0.f));
    }
    atomicAdd(&deg[d], 1.0f);
}

// ---------------------------------------------------------------------------
// Node preprocess: h = relu(relu(coords@W1+b1)@W2+b2), fp32 [N,32].
// ---------------------------------------------------------------------------
__global__ void preprocess_kernel(const float* __restrict__ coords,
                                  const float* __restrict__ w1, const float* __restrict__ b1,
                                  const float* __restrict__ w2, const float* __restrict__ b2,
                                  float* __restrict__ h, int N) {
    int n = blockIdx.x * blockDim.x + threadIdx.x;
    if (n >= N) return;
    float c0 = coords[n * 2 + 0], c1 = coords[n * 2 + 1];
    float t[16];
#pragma unroll
    for (int j = 0; j < 16; ++j)
        t[j] = fmaxf(c0 * w1[j] + c1 * w1[16 + j] + b1[j], 0.f);
#pragma unroll
    for (int o = 0; o < 32; ++o) {
        float acc = b2[o];
#pragma unroll
        for (int j = 0; j < 16; ++j) acc += t[j] * w2[j * 32 + o];
        h[(size_t)n * 32 + o] = fmaxf(acc, 0.f);
    }
}

__global__ void zero_kernel(float* __restrict__ p, int n) {
    int i = blockIdx.x * blockDim.x + threadIdx.x;
    if (i < n) p[i] = 0.f;
}

// ---------------------------------------------------------------------------
// Fused NNConv message passing (WMMA): never materializes per-edge W matrices.
// Block = 8 waves, all work on the same 16-edge batch. Wave w owns GEMM
// column tiles t = 8w..8w+7 of e3[16x32] @ en_w4[32x1024]; tile t maps to
// (i = t>>1, o-half = t&1), contributing x[src,i]*(D + b4) to the message.
// Partials combined via LDS float atomics, flushed with global f32 atomics.
// ---------------------------------------------------------------------------
__global__ __launch_bounds__(256) void msg_kernel(
        const int* __restrict__ eidx,
        const unsigned short* __restrict__ e3,
        const float* __restrict__ x,
        const float* __restrict__ w4,     // [32,1024] fp32
        const float* __restrict__ b4,     // [1024]
        float* __restrict__ agg, int E, int N) {
    const int tid  = threadIdx.x;
    const int lane = tid & 31;
    const int w    = tid >> 5;                 // wave 0..7

    __shared__ float lds_x[16 * 32];
    __shared__ float lds_msg[16 * 32];

    // Resident B fragments for this wave's 8 tiles (loaded once, reused).
    // B(32x16) layout: lane -> col n=lane&15, kb=(lane<16?0:8);
    // VGPR v holds K = (v<4 ? 2v : 16+2(v-4)) + kb, +1.
    unsigned int Breg[8][8];
    float        b4reg[8];
    {
        const int n  = lane & 15;
        const int kb = (lane < 16) ? 0 : 8;
#pragma unroll
        for (int tt = 0; tt < 8; ++tt) {
            const int t = w * 8 + tt;
            const int colbase = t * 16;
#pragma unroll
            for (int v = 0; v < 8; ++v) {
                const int k = ((v < 4) ? 2 * v : 16 + 2 * (v - 4)) + kb;
                Breg[tt][v] = pack_bf(w4[(size_t)k * 1024 + colbase + n],
                                      w4[(size_t)(k + 1) * 1024 + colbase + n]);
            }
            b4reg[tt] = b4[(t >> 1) * 32 + (t & 1) * 16 + n];
        }
    }

    const int nblk = (E + 15) >> 4;
    for (int eb = blockIdx.x; eb < nblk; eb += gridDim.x) {
        const int e0 = eb << 4;
        __syncthreads();                                  // prev flush done
        // stage x[src] rows + zero message tile
        for (int idx = tid; idx < 16 * 32; idx += 256) {
            int row = idx >> 5, c = idx & 31;
            int ee  = e0 + row; if (ee >= E) ee = E - 1;
            lds_x[idx]   = x[(size_t)eidx[ee] * 32 + c];
            lds_msg[idx] = 0.f;
        }
        // prefetch next batch's A operand while we compute
        if (eb + gridDim.x < nblk)
            __builtin_prefetch(e3 + (size_t)(e0 + (gridDim.x << 4)) * 32, 0, 1);
        __syncthreads();

        // A fragment (16 edges x K=32, bf16). A layout: lane -> row=lane&15,
        // kb=(lane<16?0:8); VGPR v: K = (v<4 ? 2v : 16+2(v-4)) + kb.
        Frag16 a;
        {
            const int row = lane & 15;
            const int kb  = (lane < 16) ? 0 : 8;
            int ee = e0 + row; if (ee >= E) ee = E - 1;
            const unsigned int* erow =
                (const unsigned int*)(e3 + (size_t)ee * 32);
#pragma unroll
            for (int v = 0; v < 8; ++v) {
                const int k0 = ((v < 4) ? 2 * v : 16 + 2 * (v - 4)) + kb;
                a.u[v] = erow[k0 >> 1];
            }
        }

        v8f acc0 = {};   // o = 0..15 contribution of this wave's 4 i-values
        v8f acc1 = {};   // o = 16..31
        const int mofs = (lane < 16) ? 0 : 8;
#pragma unroll
        for (int tt = 0; tt < 8; ++tt) {
            const int t = w * 8 + tt;
            const int i = t >> 1;
            Frag16 b;
#pragma unroll
            for (int v = 0; v < 8; ++v) b.u[v] = Breg[tt][v];
            v8f c = {};
            v8f d = wmma_bf16(a.v, b.v, c);
#pragma unroll
            for (int v = 0; v < 8; ++v) {
                float xv = lds_x[(v + mofs) * 32 + i];
                float contrib = xv * (d[v] + b4reg[tt]);
                if (tt & 1) acc1[v] += contrib; else acc0[v] += contrib;
            }
        }
        // combine the 8 waves' partial messages in LDS
        {
            const int n = lane & 15;
#pragma unroll
            for (int v = 0; v < 8; ++v) {
                atomicAdd(&lds_msg[(v + mofs) * 32 + n],      acc0[v]);
                atomicAdd(&lds_msg[(v + mofs) * 32 + 16 + n], acc1[v]);
            }
        }
        __syncthreads();
        // flush: one global f32 atomic per (edge, feature)
        for (int idx = tid; idx < 16 * 32; idx += 256) {
            int row = idx >> 5, c = idx & 31;
            int ee  = e0 + row;
            if (ee < E) {
                int dn = eidx[E + ee];
                atomicAdd(&agg[(size_t)dn * 32 + c], lds_msg[idx]);
            }
        }
    }
}

// ---------------------------------------------------------------------------
// Fused mean-agg + relu + GRU step, WMMA over 16 nodes per wave.
// gi = m @ Wih, gh = h @ Whh  (two K=32 bf16 GEMMs, 6 col tiles each).
// Gate math done directly on matching D-fragment elements.
// ---------------------------------------------------------------------------
__global__ __launch_bounds__(128) void gru_kernel(
        const float* __restrict__ agg, const float* __restrict__ deg,
        const float* __restrict__ nn_bias,
        const float* __restrict__ wih, const float* __restrict__ bih,
        const float* __restrict__ whh, const float* __restrict__ bhh,
        float* __restrict__ h, int N) {
    const int lane = threadIdx.x & 31;
    const int gw   = blockIdx.x * 4 + (threadIdx.x >> 5);
    const int n0   = gw * 16;
    if (n0 >= N) return;

    const int row = lane & 15;
    int node = n0 + row; if (node >= N) node = N - 1;
    const int kbA = (lane < 16) ? 0 : 8;
    const float rdeg = 1.0f / fmaxf(deg[node], 1.0f);

    Frag16 am, ah;
#pragma unroll
    for (int v = 0; v < 8; ++v) {
        const int k0 = ((v < 4) ? 2 * v : 16 + 2 * (v - 4)) + kbA;
        float2 ag = *(const float2*)(agg + (size_t)node * 32 + k0);
        float2 hp = *(const float2*)(h   + (size_t)node * 32 + k0);
        float m0 = fmaxf(ag.x * rdeg + nn_bias[k0],     0.f);
        float m1 = fmaxf(ag.y * rdeg + nn_bias[k0 + 1], 0.f);
        am.u[v] = pack_bf(m0, m1);
        ah.u[v] = pack_bf(hp.x, hp.y);
    }

    v8f gi[6], gh[6];
    const int n   = lane & 15;
    const int kbB = (lane < 16) ? 0 : 8;
#pragma unroll
    for (int t = 0; t < 6; ++t) {
        Frag16 bw, bh2;
#pragma unroll
        for (int v = 0; v < 8; ++v) {
            const int k = ((v < 4) ? 2 * v : 16 + 2 * (v - 4)) + kbB;
            bw.u[v]  = pack_bf(wih[(size_t)k * 96 + t * 16 + n],
                               wih[(size_t)(k + 1) * 96 + t * 16 + n]);
            bh2.u[v] = pack_bf(whh[(size_t)k * 96 + t * 16 + n],
                               whh[(size_t)(k + 1) * 96 + t * 16 + n]);
        }
        v8f c = {};
        gi[t] = wmma_bf16(am.v, bw.v, c);
        v8f c2 = {};
        gh[t] = wmma_bf16(ah.v, bh2.v, c2);
        const float bi = bih[t * 16 + n], bh = bhh[t * 16 + n];
#pragma unroll
        for (int v = 0; v < 8; ++v) { gi[t][v] += bi; gh[t][v] += bh; }
    }

    const int mofs = (lane < 16) ? 0 : 8;
#pragma unroll
    for (int th = 0; th < 2; ++th) {
#pragma unroll
        for (int v = 0; v < 8; ++v) {
            float r  = sigmoidf(gi[th][v]     + gh[th][v]);
            float zg = sigmoidf(gi[2 + th][v] + gh[2 + th][v]);
            float nc = tanhf(gi[4 + th][v] + r * gh[4 + th][v]);
            int nd = n0 + v + mofs; if (nd >= N) nd = N - 1;
            size_t addr = (size_t)nd * 32 + th * 16 + n;
            float hp = h[addr];
            h[addr] = (1.f - zg) * nc + zg * hp;      // coalesced per VGPR
        }
    }
}

// ---------------------------------------------------------------------------
// Set2Set pieces (tiny, deterministic two-level reductions).
// ---------------------------------------------------------------------------
__global__ void lstm_kernel(const float* __restrict__ wih, const float* __restrict__ bih,
                            const float* __restrict__ whh, const float* __restrict__ bhh,
                            const float* __restrict__ qstar,
                            float* __restrict__ hl, float* __restrict__ cl) {
    __shared__ float g[128];
    int j = threadIdx.x;
    float s = bih[j] + bhh[j];
    for (int k = 0; k < 64; ++k) s += qstar[k] * wih[k * 128 + j];
    for (int k = 0; k < 32; ++k) s += hl[k]    * whh[k * 128 + j];
    g[j] = s;
    __syncthreads();
    if (j < 32) {
        float i_ = g[j], f_ = g[32 + j], g_ = g[64 + j], o_ = g[96 + j];
        float c = sigmoidf(f_) * cl[j] + sigmoidf(i_) * tanhf(g_);
        cl[j] = c;
        hl[j] = sigmoidf(o_) * tanhf(c);
    }
}

__global__ __launch_bounds__(256) void attn_dot_kernel(
        const float* __restrict__ h, const float* __restrict__ hl,
        float* __restrict__ dot, float* __restrict__ bmax, int N) {
    __shared__ float shl[32];
    __shared__ float red[256];
    if (threadIdx.x < 32) shl[threadIdx.x] = hl[threadIdx.x];
    __syncthreads();
    int nidx = blockIdx.x * 256 + threadIdx.x;
    float v = -1e30f;
    if (nidx < N) {
        float s = 0.f;
#pragma unroll
        for (int i = 0; i < 32; ++i) s += h[(size_t)nidx * 32 + i] * shl[i];
        dot[nidx] = s; v = s;
    }
    red[threadIdx.x] = v;
    __syncthreads();
    for (int ofs = 128; ofs > 0; ofs >>= 1) {
        if (threadIdx.x < ofs)
            red[threadIdx.x] = fmaxf(red[threadIdx.x], red[threadIdx.x + ofs]);
        __syncthreads();
    }
    if (threadIdx.x == 0) bmax[blockIdx.x] = red[0];
}

__global__ void reduce_max_kernel(const float* __restrict__ bmax,
                                  float* __restrict__ gmax, int nb) {
    __shared__ float red[256];
    float v = -1e30f;
    for (int i = threadIdx.x; i < nb; i += 256) v = fmaxf(v, bmax[i]);
    red[threadIdx.x] = v;
    __syncthreads();
    for (int ofs = 128; ofs > 0; ofs >>= 1) {
        if (threadIdx.x < ofs)
            red[threadIdx.x] = fmaxf(red[threadIdx.x], red[threadIdx.x + ofs]);
        __syncthreads();
    }
    if (threadIdx.x == 0) gmax[0] = red[0];
}

__global__ __launch_bounds__(256) void attn_acc_kernel(
        const float* __restrict__ h, const float* __restrict__ dot,
        const float* __restrict__ gmax,
        float* __restrict__ bsum, float* __restrict__ bvec, int N) {
    __shared__ float ev[256];
    __shared__ float red[256];
    const int nb = blockIdx.x * 256;
    const int nidx = nb + threadIdx.x;
    float e = 0.f;
    if (nidx < N) e = __expf(dot[nidx] - gmax[0]);
    ev[threadIdx.x] = e;
    red[threadIdx.x] = e;
    __syncthreads();
    for (int ofs = 128; ofs > 0; ofs >>= 1) {
        if (threadIdx.x < ofs) red[threadIdx.x] += red[threadIdx.x + ofs];
        __syncthreads();
    }
    if (threadIdx.x == 0) bsum[blockIdx.x] = red[0];
    __syncthreads();
    // weighted column sums (coalesced over features)
    const int o = threadIdx.x & 31, g = threadIdx.x >> 5;
    int lim = N - nb; if (lim > 256) lim = 256;
    float acc = 0.f;
    for (int j = g; j < lim; j += 8) acc += ev[j] * h[(size_t)(nb + j) * 32 + o];
    red[g * 32 + o] = acc;
    __syncthreads();
    if (g == 0) {
        float a = 0.f;
#pragma unroll
        for (int q = 0; q < 8; ++q) a += red[q * 32 + o];
        bvec[(size_t)blockIdx.x * 32 + o] = a;
    }
}

__global__ void attn_final_kernel(const float* __restrict__ bsum,
                                  const float* __restrict__ bvec,
                                  const float* __restrict__ hl,
                                  float* __restrict__ qstar, int nb) {
    __shared__ float stot;
    if (threadIdx.x == 0) {
        float s = 0.f;
        for (int i = 0; i < nb; ++i) s += bsum[i];
        stot = s;
    }
    __syncthreads();
    if (threadIdx.x < 32) {
        float a = 0.f;
        for (int b = 0; b < nb; ++b) a += bvec[(size_t)b * 32 + threadIdx.x];
        qstar[threadIdx.x]      = hl[threadIdx.x];
        qstar[32 + threadIdx.x] = a / stot;
    }
}

__global__ __launch_bounds__(512) void mlp_kernel(
        const float* __restrict__ qstar,
        const float* __restrict__ w1, const float* __restrict__ b1,
        const float* __restrict__ w2, const float* __restrict__ b2,
        float* __restrict__ out) {
    __shared__ float q[64];
    __shared__ float red[512];
    int j = threadIdx.x;
    if (j < 64) q[j] = qstar[j];
    __syncthreads();
    float s = b1[j];
    for (int k = 0; k < 64; ++k) s += q[k] * w1[(size_t)k * 512 + j];
    red[j] = fmaxf(s, 0.f) * w2[j];
    __syncthreads();
    for (int ofs = 256; ofs > 0; ofs >>= 1) {
        if (j < ofs) red[j] += red[j + ofs];
        __syncthreads();
    }
    if (j == 0) out[0] = red[0] + b2[0];
}

// ---------------------------------------------------------------------------
extern "C" void kernel_launch(void* const* d_in, const int* in_sizes, int n_in,
                              void* d_out, int out_size, void* d_ws, size_t ws_size,
                              hipStream_t stream) {
    const float* coords   = (const float*)d_in[0];
    const int*   eidx     = (const int*)  d_in[1];
    const float* pre_w1   = (const float*)d_in[2];
    const float* pre_b1   = (const float*)d_in[3];
    const float* pre_w2   = (const float*)d_in[4];
    const float* pre_b2   = (const float*)d_in[5];
    const float* en_w1    = (const float*)d_in[6];
    const float* en_b1    = (const float*)d_in[7];
    const float* en_w2    = (const float*)d_in[8];
    const float* en_b2    = (const float*)d_in[9];
    const float* en_w3    = (const float*)d_in[10];
    const float* en_b3    = (const float*)d_in[11];
    const float* en_w4    = (const float*)d_in[12];
    const float* en_b4    = (const float*)d_in[13];
    const float* nn_bias  = (const float*)d_in[14];
    const float* gru_wih  = (const float*)d_in[15];
    const float* gru_bih  = (const float*)d_in[16];
    const float* gru_whh  = (const float*)d_in[17];
    const float* gru_bhh  = (const float*)d_in[18];
    const float* lstm_wih = (const float*)d_in[19];
    const float* lstm_bih = (const float*)d_in[20];
    const float* lstm_whh = (const float*)d_in[21];
    const float* lstm_bhh = (const float*)d_in[22];
    const float* pr_w1    = (const float*)d_in[23];
    const float* pr_b1    = (const float*)d_in[24];
    const float* pr_w2    = (const float*)d_in[25];
    const float* pr_b2    = (const float*)d_in[26];
    float* out = (float*)d_out;

    const int N = in_sizes[0] / 2;
    const int E = in_sizes[1] / 2;
    if (N <= 0 || E <= 0) return;
    const int nb1 = (N + 255) / 256;

    // workspace layout (floats)
    float* ws = (float*)d_ws;
    size_t off = 0;
    float* h     = ws + off; off += (size_t)N * 32;
    float* agg   = ws + off; off += (size_t)N * 32;
    float* deg   = ws + off; off += N;
    float* dotb  = ws + off; off += N;
    float* bmax  = ws + off; off += nb1;
    float* bsum  = ws + off; off += nb1;
    float* bvec  = ws + off; off += (size_t)nb1 * 32;
    float* gmax  = ws + off; off += 1;
    float* hl    = ws + off; off += 32;   // hl, cl, qstar contiguous (zeroed together)
    float* cl    = ws + off; off += 32;
    float* qstar = ws + off; off += 64;
    unsigned short* e3 = (unsigned short*)(ws + off);

    // ---- init ----
    zero_kernel<<<(N + 255) / 256, 256, 0, stream>>>(deg, N);
    zero_kernel<<<1, 128, 0, stream>>>(hl, 128);   // hl+cl+qstar

    edgenet_kernel<<<(E + 255) / 256, 256, 0, stream>>>(
        eidx, coords, en_w1, en_b1, en_w2, en_b2, en_w3, en_b3, e3, deg, E);
    preprocess_kernel<<<(N + 255) / 256, 256, 0, stream>>>(
        coords, pre_w1, pre_b1, pre_w2, pre_b2, h, N);

    // ---- 3 rounds of fused NNConv + GRU ----
    const int nblk = (E + 15) >> 4;
    int mblocks = nblk < 2048 ? nblk : 2048;
    const int gru_blocks = ((N + 15) / 16 + 3) / 4;
    for (int it = 0; it < 3; ++it) {
        zero_kernel<<<((size_t)N * 32 + 255) / 256, 256, 0, stream>>>(agg, N * 32);
        msg_kernel<<<mblocks, 256, 0, stream>>>(eidx, e3, h, en_w4, en_b4, agg, E, N);
        gru_kernel<<<gru_blocks, 128, 0, stream>>>(
            agg, deg, nn_bias, gru_wih, gru_bih, gru_whh, gru_bhh, h, N);
    }

    // ---- Set2Set (6 iterations) ----
    for (int s = 0; s < 6; ++s) {
        lstm_kernel<<<1, 128, 0, stream>>>(lstm_wih, lstm_bih, lstm_whh, lstm_bhh,
                                           qstar, hl, cl);
        attn_dot_kernel<<<nb1, 256, 0, stream>>>(h, hl, dotb, bmax, N);
        reduce_max_kernel<<<1, 256, 0, stream>>>(bmax, gmax, nb1);
        attn_acc_kernel<<<nb1, 256, 0, stream>>>(h, dotb, gmax, bsum, bvec, N);
        attn_final_kernel<<<1, 64, 0, stream>>>(bsum, bvec, hl, qstar, nb1);
    }

    // ---- readout MLP ----
    mlp_kernel<<<1, 512, 0, stream>>>(qstar, pr_w1, pr_b1, pr_w2, pr_b2, out);
}
